// LlamaSdpaAttention_60962765799573
// MI455X (gfx1250) — compile-verified, matrix operand order checked by reading, not run on hardware
//
#include <hip/hip_runtime.h>
#include <hip/hip_fp16.h>

typedef __attribute__((ext_vector_type(8)))  _Float16 v8h;
typedef __attribute__((ext_vector_type(16))) _Float16 v16h;
typedef __attribute__((ext_vector_type(8)))  float    v8f;

#define T_SEQ 2048
#define NEMBD 2048
#define NH    32
#define NKV   8
#define HD    64
#define BATCH 2
#define M_ROWS (BATCH * T_SEQ)   // 4096 token rows

// ---------------------------------------------------------------------------
// WMMA operand loader.
// A (16x32 f16, row-major M x K) per CDNA5 ISA:
//   lane%16 = row; lane/16 = half; v16h elems [0..7]  = K (half*8 + i)
//                                  elems [8..15] = K (16 + half*8 + i)
// B uses the identical pattern over row-major N x K data (lane%16 = column).
// Both are two 16-byte loads per lane.
// ---------------------------------------------------------------------------
static __device__ __forceinline__ v16h load_ab16(const _Float16* row, int half8) {
    v8h lo = *(const v8h*)(row + half8);
    v8h hi = *(const v8h*)(row + 16 + half8);
    v16h r;
#pragma unroll
    for (int i = 0; i < 8; ++i) { r[i] = lo[i]; r[8 + i] = hi[i]; }
    return r;
}

static __device__ __forceinline__ v8f wmma_f16(v16h a, v16h b, v8f c) {
    return __builtin_amdgcn_wmma_f32_16x16x32_f16(false, a, false, b,
                                                  (short)0, c, false, false);
}

// ---------------------------------------------------------------------------
// f32 -> f16 conversion (one-time; weights+x fit in the 192MB L2 afterwards)
// ---------------------------------------------------------------------------
__global__ void f32_to_f16_kernel(const float* __restrict__ src,
                                  _Float16* __restrict__ dst, int n) {
    int i = blockIdx.x * blockDim.x + threadIdx.x;
    int stride = gridDim.x * blockDim.x;
    for (; i < n; i += stride) dst[i] = (_Float16)src[i];
}

// ---------------------------------------------------------------------------
// Y = X * W^T WMMA GEMM, register-blocked 64x64 per wave32.
// X: M_ROWS x 2048 f16.  W: N x 2048 f16 (row-major (out,in) = WMMA B layout).
// Inner loop: 4 B operands stay resident; one A operand at a time feeds 4
// WMMAs -> peak live ~= 128 acc + 32 B + 8 A VGPRs.  __launch_bounds__(128,1)
// raises the VGPR budget (gfx1250: up to 1024/wave) so nothing spills.
// mode 0: Q proj  -> RoPE -> Qh [b][h][t][d]        (f16)
// mode 1: K proj  -> RoPE -> Kh [b][kvh][t][d]      (f16)
// mode 2: V proj  ->         Vt [b][kvh][d][t]      (f16, transposed for PV)
// mode 3: out proj ->        d_out [b][t][c]        (f32)
// ---------------------------------------------------------------------------
__global__ void __launch_bounds__(128, 1)
gemm_wmma_kernel(const _Float16* __restrict__ X,
                 const _Float16* __restrict__ W,
                 void* __restrict__ dst,
                 const float* __restrict__ fcos,
                 const float* __restrict__ fsin,
                 int N, int mode) {
    const int lane   = threadIdx.x & 31;
    const int wave   = threadIdx.x >> 5;
    const int tile   = blockIdx.x * 4 + wave;
    const int ntn    = N >> 6;                 // 64-wide n tiles
    const int ntiles = (M_ROWS >> 6) * ntn;
    if (tile >= ntiles) return;

    const int m0    = (tile / ntn) << 6;
    const int n0    = (tile % ntn) << 6;
    const int lm    = lane & 15;
    const int half8 = (lane >> 4) << 3;

    const _Float16* xrow = X + (size_t)(m0 + lm) * NEMBD;
    const _Float16* wrow = W + (size_t)(n0 + lm) * NEMBD;

    v8f acc[4][4] = {};                        // [mi][ni], 128 VGPRs
    for (int kk = 0; kk < NEMBD; kk += 32) {
        v16h b[4];
#pragma unroll
        for (int ni = 0; ni < 4; ++ni)
            b[ni] = load_ab16(wrow + (size_t)ni * 16 * NEMBD + kk, half8);
#pragma unroll
        for (int mi = 0; mi < 4; ++mi) {
            v16h a = load_ab16(xrow + (size_t)mi * 16 * NEMBD + kk, half8);
#pragma unroll
            for (int ni = 0; ni < 4; ++ni)
                acc[mi][ni] = wmma_f16(a, b[ni], acc[mi][ni]);
        }
    }

    if (mode == 3) {
        float* out = (float*)dst;
#pragma unroll
        for (int mi = 0; mi < 4; ++mi)
#pragma unroll
            for (int ni = 0; ni < 4; ++ni) {
                int n = n0 + ni * 16 + lm;
#pragma unroll
                for (int r = 0; r < 8; ++r) {
                    int m = m0 + mi * 16 + r + half8;   // C: row = r + 8*half
                    out[(size_t)m * NEMBD + n] = acc[mi][ni][r];
                }
            }
        return;
    }

    _Float16* out16 = (_Float16*)dst;
#pragma unroll
    for (int mi = 0; mi < 4; ++mi)
#pragma unroll
        for (int ni = 0; ni < 4; ++ni) {
            int n  = n0 + ni * 16 + lm;        // output feature of this lane
            int h  = n >> 6;                   // head index (32 or 8 heads)
            int d  = n & (HD - 1);
            int fi = d >> 1;                   // rope frequency index
#pragma unroll
            for (int r = 0; r < 8; ++r) {
                int m    = m0 + mi * 16 + r + half8;
                int bidx = m >> 11;            // / T_SEQ
                int t    = m & (T_SEQ - 1);
                float v  = acc[mi][ni][r];
                if (mode != 2) {               // RoPE: pairs = adjacent lanes
                    float c     = fcos[t * (HD / 2) + fi];
                    float s     = fsin[t * (HD / 2) + fi];
                    float other = __shfl_xor(v, 1, 32);
                    v = (d & 1) ? (other * s + v * c)   // odd  = a*s + b*c
                                : (v * c - other * s);  // even = a*c - b*s
                }
                size_t off;
                if (mode == 0)
                    off = ((size_t)(bidx * NH  + h) * T_SEQ + t) * HD + d;
                else if (mode == 1)
                    off = ((size_t)(bidx * NKV + h) * T_SEQ + t) * HD + d;
                else // mode 2: V transposed -> [b][kvh][d][t]
                    off = ((size_t)(bidx * NKV + h) * HD + d) * T_SEQ + t;
                out16[off] = (_Float16)v;
            }
        }
}

// ---------------------------------------------------------------------------
// Flash attention: one wave32 owns 32 query rows (2 m-subtiles) of one
// (b, head).  Per causal s-chunk of 32: 8 score WMMAs + online softmax
// (shfl_xor reductions within 16-lane half-groups), P transposed to WMMA A
// layout via per-wave LDS bounce (ds ops + s_wait_dscnt), 8 PV WMMAs into
// 32x64 f32 accumulators.  16 K/V loads feed 16 WMMAs per chunk.
// ---------------------------------------------------------------------------
__global__ void __launch_bounds__(128, 1)
flash_attn_kernel(const _Float16* __restrict__ Qh,
                  const _Float16* __restrict__ Kh,
                  const _Float16* __restrict__ Vt,
                  _Float16* __restrict__ attn_out) {
    __shared__ _Float16 plds[4][32][32];       // per-wave 32x32 P tile (8KB)

    const int lane  = threadIdx.x & 31;
    const int wave  = threadIdx.x >> 5;
    const int g     = blockIdx.x * 4 + wave;
    const int bh    = g >> 6;                  // / (T/32 = 64)
    const int tq    = g & 63;
    const int t0    = tq << 5;                 // 32 q rows per wave
    const int b     = bh >> 5;
    const int h     = bh & 31;
    const int kvh   = h >> 2;                  // GQA 4:1
    const int lm    = lane & 15;
    const int half8 = (lane >> 4) << 3;

    const _Float16* Qp = Qh + ((size_t)(b * NH  + h)   * T_SEQ + t0) * HD;
    const _Float16* Kp = Kh +  (size_t)(b * NKV + kvh) * T_SEQ * HD;
    const _Float16* Vp = Vt +  (size_t)(b * NKV + kvh) * HD * T_SEQ;

    // Q tile (32 x 64) in registers: [m-subtile][k-step]
    v16h qa[2][2];
#pragma unroll
    for (int mi = 0; mi < 2; ++mi) {
        qa[mi][0] = load_ab16(Qp + (size_t)(mi * 16 + lm) * HD,      half8);
        qa[mi][1] = load_ab16(Qp + (size_t)(mi * 16 + lm) * HD + 32, half8);
    }

    v8f o[2][4] = {};                          // [m-subtile][d-tile]
    float mrow[2][8], lrow[2][8];
#pragma unroll
    for (int mi = 0; mi < 2; ++mi)
#pragma unroll
        for (int r = 0; r < 8; ++r) { mrow[mi][r] = -3.0e38f; lrow[mi][r] = 0.0f; }

    const float scale = 0.125f;                // 1/sqrt(64)

    for (int s0 = 0; s0 < t0 + 32; s0 += 32) {
        // ---- K operands for both 16-wide s-subtiles -----------------------
        v16h ka[2][2];
#pragma unroll
        for (int si = 0; si < 2; ++si) {
            ka[si][0] = load_ab16(Kp + (size_t)(s0 + si * 16 + lm) * HD,      half8);
            ka[si][1] = load_ab16(Kp + (size_t)(s0 + si * 16 + lm) * HD + 32, half8);
        }

        // ---- scores: S[mi][si] = Q K^T -----------------------------------
        v8f S[2][2];
#pragma unroll
        for (int mi = 0; mi < 2; ++mi)
#pragma unroll
            for (int si = 0; si < 2; ++si) {
                v8f s = {};
                s = wmma_f16(qa[mi][0], ka[si][0], s);
                s = wmma_f16(qa[mi][1], ka[si][1], s);
                S[mi][si] = s;
            }

        // ---- causal mask + online softmax per m-subtile ------------------
#pragma unroll
        for (int mi = 0; mi < 2; ++mi) {
#pragma unroll
            for (int r = 0; r < 8; ++r) {
                int tcur = t0 + mi * 16 + r + half8;
                int sa   = s0 + lm;
                int sb   = s0 + 16 + lm;
                float va = S[mi][0][r] * scale + (sa > tcur ? -1.0e9f : 0.0f);
                float vb = S[mi][1][r] * scale + (sb > tcur ? -1.0e9f : 0.0f);

                float mx = fmaxf(va, vb);      // row reduce over 16 lanes
                mx = fmaxf(mx, __shfl_xor(mx, 1, 32));
                mx = fmaxf(mx, __shfl_xor(mx, 2, 32));
                mx = fmaxf(mx, __shfl_xor(mx, 4, 32));
                mx = fmaxf(mx, __shfl_xor(mx, 8, 32));

                float mnew = fmaxf(mrow[mi][r], mx);
                float cr   = __expf(mrow[mi][r] - mnew);
                mrow[mi][r] = mnew;

                float pa = __expf(va - mnew);
                float pb = __expf(vb - mnew);
                float rs = pa + pb;
                rs += __shfl_xor(rs, 1, 32);
                rs += __shfl_xor(rs, 2, 32);
                rs += __shfl_xor(rs, 4, 32);
                rs += __shfl_xor(rs, 8, 32);
                lrow[mi][r] = lrow[mi][r] * cr + rs;

                // stash P (C layout) into LDS for transpose to A layout
                plds[wave][mi * 16 + r + half8][lm]      = (_Float16)pa;
                plds[wave][mi * 16 + r + half8][16 + lm] = (_Float16)pb;

                o[mi][0][r] *= cr; o[mi][1][r] *= cr;
                o[mi][2][r] *= cr; o[mi][3][r] *= cr;
            }
        }

        // wave-private LDS bounce: wait for ds stores before reading back
        asm volatile("s_wait_dscnt 0" ::: "memory");
        v16h pm[2];
#pragma unroll
        for (int mi = 0; mi < 2; ++mi)
            pm[mi] = load_ab16(&plds[wave][mi * 16 + lm][0], half8);

        // ---- O += P * V (V transposed: rows = d, cols = s) ---------------
        v16h vb[4];
#pragma unroll
        for (int j = 0; j < 4; ++j)
            vb[j] = load_ab16(Vp + (size_t)(j * 16 + lm) * T_SEQ + s0, half8);
#pragma unroll
        for (int mi = 0; mi < 2; ++mi)
#pragma unroll
            for (int j = 0; j < 4; ++j)
                o[mi][j] = wmma_f16(pm[mi], vb[j], o[mi][j]);
    }

    // ---- normalize and write [b][t][h*64+d] as f16 for the out-proj ------
#pragma unroll
    for (int mi = 0; mi < 2; ++mi)
#pragma unroll
        for (int r = 0; r < 8; ++r) {
            float inv = 1.0f / lrow[mi][r];
            int t = t0 + mi * 16 + r + half8;
            size_t base = ((size_t)b * T_SEQ + t) * NEMBD + (size_t)h * HD;
            attn_out[base +  0 + lm] = (_Float16)(o[mi][0][r] * inv);
            attn_out[base + 16 + lm] = (_Float16)(o[mi][1][r] * inv);
            attn_out[base + 32 + lm] = (_Float16)(o[mi][2][r] * inv);
            attn_out[base + 48 + lm] = (_Float16)(o[mi][3][r] * inv);
        }
}

// ---------------------------------------------------------------------------
// inputs: x, freqs_cos, freqs_sin, attention_mask(unused; causal inline),
//         Wq, Wk, Wv, Wo   (all f32).  output: (B,T,2048) f32.
// ---------------------------------------------------------------------------
extern "C" void kernel_launch(void* const* d_in, const int* in_sizes, int n_in,
                              void* d_out, int out_size, void* d_ws, size_t ws_size,
                              hipStream_t stream) {
    const float* x    = (const float*)d_in[0];
    const float* fcos = (const float*)d_in[1];
    const float* fsin = (const float*)d_in[2];
    const float* Wq   = (const float*)d_in[4];
    const float* Wk   = (const float*)d_in[5];
    const float* Wv   = (const float*)d_in[6];
    const float* Wo   = (const float*)d_in[7];

    _Float16* ws = (_Float16*)d_ws;
    size_t off = 0;
    _Float16* x16  = ws + off; off += (size_t)M_ROWS * NEMBD;            // 8M
    _Float16* wq16 = ws + off; off += (size_t)NEMBD * NEMBD;             // 4M
    _Float16* wk16 = ws + off; off += (size_t)NKV * HD * NEMBD;          // 1M
    _Float16* wv16 = ws + off; off += (size_t)NKV * HD * NEMBD;          // 1M
    _Float16* wo16 = ws + off; off += (size_t)NEMBD * NEMBD;             // 4M
    _Float16* qh   = ws + off; off += (size_t)BATCH * NH  * T_SEQ * HD;  // 8M
    _Float16* kh   = ws + off; off += (size_t)BATCH * NKV * T_SEQ * HD;  // 2M
    _Float16* vt   = ws + off; off += (size_t)BATCH * NKV * HD * T_SEQ;  // 2M
    _Float16* at16 = ws + off; off += (size_t)M_ROWS * NEMBD;            // 8M

    // 1) precision drop to f16
    f32_to_f16_kernel<<<2048, 256, 0, stream>>>(x,  x16,  M_ROWS * NEMBD);
    f32_to_f16_kernel<<<2048, 256, 0, stream>>>(Wq, wq16, NEMBD * NEMBD);
    f32_to_f16_kernel<<<1024, 256, 0, stream>>>(Wk, wk16, NKV * HD * NEMBD);
    f32_to_f16_kernel<<<1024, 256, 0, stream>>>(Wv, wv16, NKV * HD * NEMBD);
    f32_to_f16_kernel<<<2048, 256, 0, stream>>>(Wo, wo16, NEMBD * NEMBD);

    // 2) QKV projections (+RoPE, +V transpose); 64x64 tiles per wave
    //    Q: 64*32 = 2048 tiles -> 512 blocks; K/V: 64*8 = 512 tiles -> 128
    gemm_wmma_kernel<<<512, 128, 0, stream>>>(x16, wq16, qh, fcos, fsin, NEMBD,    0);
    gemm_wmma_kernel<<<128, 128, 0, stream>>>(x16, wk16, kh, fcos, fsin, NKV * HD, 1);
    gemm_wmma_kernel<<<128, 128, 0, stream>>>(x16, wv16, vt, fcos, fsin, NKV * HD, 2);

    // 3) fused causal flash attention: 64 bh * 64 q-tiles / 4 waves = 1024
    flash_attn_kernel<<<1024, 128, 0, stream>>>(qh, kh, vt, at16);

    // 4) output projection, f32 result straight to d_out
    gemm_wmma_kernel<<<512, 128, 0, stream>>>(at16, wo16, d_out, fcos, fsin, NEMBD, 3);
}
